// Seq2SeqLSTM_62491774156957
// MI455X (gfx1250) — compile-verified, matrix operand order checked by reading, not run on hardware
//
#include <hip/hip_runtime.h>

// ---------------- problem dims ----------------
#define B_    64
#define SSRC  128
#define STGT  64
#define EMB_  128
#define HID_  256
#define G4    1024   // 4*HID
#define VTGT  64

// ---------------- WMMA types ----------------
typedef __bf16 bf16_t;
typedef bf16_t v16bf __attribute__((ext_vector_type(16)));
typedef float  v8f   __attribute__((ext_vector_type(8)));

union FragU { unsigned int u[8]; v16bf v; };

__device__ __forceinline__ unsigned short f2bf(float f) {
  union { float f; unsigned int u; } x; x.f = f;
  unsigned int u = x.u;
  unsigned int r = (u + 0x7FFFu + ((u >> 16) & 1u)) >> 16;   // RNE
  return (unsigned short)r;
}
__device__ __forceinline__ float bf2f(unsigned short u) {
  union { unsigned int u; float f; } x; x.u = ((unsigned int)u) << 16;
  return x.f;
}
__device__ __forceinline__ float sigf(float x) { return 1.0f / (1.0f + __expf(-x)); }

// load one 16x32 bf16 fragment (A-layout; also used for W stored as [N][K])
__device__ __forceinline__ v16bf ldfrag(const unsigned short* p) {
  FragU f;
#pragma unroll
  for (int g = 0; g < 4; ++g) {
    f.u[g]     = *(const unsigned int*)(p + 2 * g);
    f.u[4 + g] = *(const unsigned int*)(p + 16 + 2 * g);
  }
  return f.v;
}

// core: acc[4] (16x64 tile) over K; A row-base / W row-base pre-offset by lane
__device__ __forceinline__ void gemm_core(const unsigned short* __restrict__ Abase,
                                          const unsigned short* __restrict__ Wb,
                                          int K, v8f acc[4]) {
  for (int k0 = 0; k0 < K; k0 += 32) {
    v16bf fa = ldfrag(Abase + k0);
    v16bf w0 = ldfrag(Wb + k0);
    v16bf w1 = ldfrag(Wb + 16 * K + k0);
    v16bf w2 = ldfrag(Wb + 32 * K + k0);
    v16bf w3 = ldfrag(Wb + 48 * K + k0);
    acc[0] = __builtin_amdgcn_wmma_f32_16x16x32_bf16(false, fa, false, w0, (short)0, acc[0], false, false);
    acc[1] = __builtin_amdgcn_wmma_f32_16x16x32_bf16(false, fa, false, w1, (short)0, acc[1], false, false);
    acc[2] = __builtin_amdgcn_wmma_f32_16x16x32_bf16(false, fa, false, w2, (short)0, acc[2], false, false);
    acc[3] = __builtin_amdgcn_wmma_f32_16x16x32_bf16(false, fa, false, w3, (short)0, acc[3], false, false);
  }
}

// ---------------- plain GEMM: C = A*W^T (+Cin) (+bias, optionally gate-permuted index) ----
__global__ __launch_bounds__(256)
void k_gemm(const unsigned short* __restrict__ A, int lda,
            const unsigned short* __restrict__ W,
            const float* __restrict__ Cin, int ldcin,
            float* __restrict__ Cout, int ldc,
            const float* __restrict__ bias, int permBias,
            int M, int N, int K)
{
  const int lane = threadIdx.x & 31;
  const int wid  = (blockIdx.x * blockDim.x + threadIdx.x) >> 5;
  const int Mt = M >> 4, Nt = N >> 6;
  if (wid >= Mt * Nt) return;
  const int mt = wid % Mt, ng = wid / Mt;
  const int row = lane & 15, half8 = (lane >> 4) << 3;

  v8f acc[4];
#pragma unroll
  for (int j = 0; j < 4; ++j) {
    if (Cin) {
#pragma unroll
      for (int r = 0; r < 8; ++r)
        acc[j][r] = Cin[((mt << 4) + r + half8) * ldcin + (ng << 6) + (j << 4) + row];
    } else {
#pragma unroll
      for (int r = 0; r < 8; ++r) acc[j][r] = 0.f;
    }
  }

  const unsigned short* Abase = A + ((mt << 4) + row) * lda + half8;
  const unsigned short* Wb    = W + ((ng << 6) + row) * K + half8;
  gemm_core(Abase, Wb, K, acc);

#pragma unroll
  for (int j = 0; j < 4; ++j) {
    int n = (ng << 6) + (j << 4) + row;
    float bv = 0.f;
    if (bias) bv = permBias ? bias[((n & 3) << 8) | (n >> 2)] : bias[n];
#pragma unroll
    for (int r = 0; r < 8; ++r)
      Cout[((mt << 4) + r + half8) * ldc + n] = acc[j][r] + bv;
  }
}

// ---------------- fused GEMM + LSTM cell (gate-interleaved weight layout) ----------------
// M=64, N=1024 fixed. Column n' = 4*h + gate. Epilogue gathers the 4 gates of each
// (m,h) from the 4 adjacent lanes via __shfl (ds_bpermute) and applies the cell.
__global__ __launch_bounds__(256)
void k_gemm_lstm(const unsigned short* __restrict__ A, int lda,
                 const unsigned short* __restrict__ W,
                 const float* __restrict__ Cin, int ldcin,   // permuted layout (or null)
                 const float* __restrict__ bias,             // original layout (or null)
                 float* __restrict__ c,                      // [64,256] in place
                 unsigned short* __restrict__ h0t, int ld0,
                 unsigned short* __restrict__ h1t, int ld1,
                 float* __restrict__ hft, int ldf,
                 int K)
{
  const int lane = threadIdx.x & 31;
  const int wid  = (blockIdx.x * blockDim.x + threadIdx.x) >> 5;
  const int Mt = 4, Nt = 16;
  if (wid >= Mt * Nt) return;
  const int mt = wid % Mt, ng = wid / Mt;
  const int row = lane & 15, half8 = (lane >> 4) << 3;

  v8f acc[4];
#pragma unroll
  for (int j = 0; j < 4; ++j) {
    if (Cin) {
#pragma unroll
      for (int r = 0; r < 8; ++r)
        acc[j][r] = Cin[((mt << 4) + r + half8) * ldcin + (ng << 6) + (j << 4) + row];
    } else {
#pragma unroll
      for (int r = 0; r < 8; ++r) acc[j][r] = 0.f;
    }
  }

  const unsigned short* Abase = A + ((mt << 4) + row) * lda + half8;
  const unsigned short* Wb    = W + ((ng << 6) + row) * K + half8;
  gemm_core(Abase, Wb, K, acc);

  const int baseLane = lane & ~3;          // 4-lane gate group
  const int q = (lane & 15) >> 2;          // h sub-index within tile
#pragma unroll
  for (int j = 0; j < 4; ++j) {
#pragma unroll
    for (int r = 0; r < 8; ++r) {
      float val = acc[j][r];
      if (bias) {
        int n = (ng << 6) + (j << 4) + (lane & 15);
        val += bias[((n & 3) << 8) | (n >> 2)];
      }
      float vi = __shfl(val, baseLane + 0, 32);
      float vf = __shfl(val, baseLane + 1, 32);
      float vg = __shfl(val, baseLane + 2, 32);
      float vo = __shfl(val, baseLane + 3, 32);
      if ((lane & 3) == 0) {
        int m = (mt << 4) + r + half8;
        int h = (ng << 4) + (j << 2) + q;
        float cn = sigf(vf) * c[m * HID_ + h] + sigf(vi) * tanhf(vg);
        c[m * HID_ + h] = cn;
        float hn = sigf(vo) * tanhf(cn);
        unsigned short hb = f2bf(hn);
        h0t[m * ld0 + h] = hb;
        if (h1t) h1t[m * ld1 + h] = hb;
        if (hft) hft[m * ldf + h] = hn;
      }
    }
  }
}

// ---------------- weight convert+transpose, optional gate interleave ---------------------
// f32 [K,N] -> bf16 [N][ldk] (+koff). perm: output col n reads orig col ((n&3)<<8)|(n>>2).
__global__ void k_convT(const float* __restrict__ W, unsigned short* __restrict__ dstT,
                        int K, int N, int ldk, int koff, int perm)
{
  int i = blockIdx.x * blockDim.x + threadIdx.x;
  if (i >= K * N) return;
  int k = i / N, n = i % N;
  int on = perm ? (((n & 3) << 8) | (n >> 2)) : n;
  dstT[n * ldk + koff + k] = f2bf(W[k * N + on]);
}

__global__ void k_zero(unsigned int* __restrict__ p, int n)
{
  int i = blockIdx.x * blockDim.x + threadIdx.x;
  if (i < n) p[i] = 0u;
}

// encoder embedding gather: rows m = s*B + b, bf16 [8192,128]
__global__ void k_gather_src(const int* __restrict__ src, const float* __restrict__ emb,
                             unsigned short* __restrict__ out)
{
  int i = blockIdx.x * blockDim.x + threadIdx.x;
  if (i >= SSRC * B_ * EMB_) return;
  int e = i % EMB_;
  int m = i / EMB_;
  int b = m % B_;
  int s = m / B_;
  int tok = src[b * SSRC + s];
  out[m * EMB_ + e] = f2bf(emb[tok * EMB_ + e]);
}

// hand encoder-final h (bf16) to decoder concat buffers
__global__ void k_copy_dec_init(const unsigned short* __restrict__ hcatE,
                                unsigned short* __restrict__ hcatD,
                                unsigned short* __restrict__ inp0,
                                unsigned short* __restrict__ logitsA)
{
  int i = blockIdx.x * blockDim.x + threadIdx.x;
  if (i >= 4 * B_ * HID_) return;
  int h = i % HID_;
  int b = (i / HID_) % B_;
  int which = i / (B_ * HID_);
  if (which == 0)      hcatD[b * 512 + h]        = hcatE[b * 512 + h];
  else if (which == 1) inp0[b * 640 + 384 + h]   = hcatE[b * 512 + h];
  else if (which == 2) hcatD[b * 512 + 256 + h]  = hcatE[b * 512 + 256 + h];
  else                 logitsA[b * 512 + h]      = hcatE[b * 512 + 256 + h];
}

// ---------------- fused q-GEMM + attention + decoder embedding pack ----------------------
__global__ __launch_bounds__(128)
void k_attn_fused(const unsigned short* __restrict__ hcatD,   // [64,512], h1 at col 256
                  const unsigned short* __restrict__ WaT,     // [256][256] bf16
                  const float* __restrict__ Uh, const float* __restrict__ v,
                  const float* __restrict__ enc_out,
                  const int* __restrict__ tgt, const float* __restrict__ dec_emb, int t,
                  unsigned short* __restrict__ ctx0, int ld0,
                  unsigned short* __restrict__ ctx1, int ld1,
                  unsigned short* __restrict__ embDst)        // inp0 base, cols 0..127
{
  __shared__ float sh1[HID_], sq[HID_], sv[HID_], sw[SSRC], red[SSRC];
  int b = blockIdx.x, tid = threadIdx.x;     // 128 threads
  sh1[tid]       = bf2f(hcatD[b * 512 + 256 + tid]);
  sh1[tid + 128] = bf2f(hcatD[b * 512 + 256 + tid + 128]);
  sv[tid] = v[tid]; sv[tid + 128] = v[tid + 128];
  // decoder embedding pack (independent)
  {
    int tok = tgt[b * STGT + t];
    embDst[b * 640 + tid] = f2bf(dec_emb[tok * EMB_ + tid]);
  }
  __syncthreads();
  // q = h1 @ Wa (2 outputs per thread)
  for (int hh = tid; hh < HID_; hh += 128) {
    const unsigned short* wr = WaT + hh * HID_;
    float a = 0.f;
    for (int k = 0; k < HID_; ++k) a += sh1[k] * bf2f(wr[k]);
    sq[hh] = a;
  }
  __syncthreads();
  // score_s = v . tanh(q + Uh[b,s,:])
  const float* uh = Uh + (b * SSRC + tid) * HID_;
  float sc = 0.f;
  for (int h = 0; h < HID_; ++h) sc += sv[h] * tanhf(sq[h] + uh[h]);
  red[tid] = sc; __syncthreads();
  for (int off = 64; off > 0; off >>= 1) { if (tid < off) red[tid] = fmaxf(red[tid], red[tid + off]); __syncthreads(); }
  float mx = red[0]; __syncthreads();
  float e = __expf(sc - mx);
  sw[tid] = e; red[tid] = e; __syncthreads();
  for (int off = 64; off > 0; off >>= 1) { if (tid < off) red[tid] += red[tid + off]; __syncthreads(); }
  float inv = 1.f / red[0];
  // ctx = softmax(score) . enc_out
  const float* eo = enc_out + b * SSRC * HID_;
  float c0 = 0.f, c1 = 0.f;
  for (int s = 0; s < SSRC; ++s) {
    float w = sw[s];
    c0 += w * eo[s * HID_ + tid];
    c1 += w * eo[s * HID_ + tid + 128];
  }
  c0 *= inv; c1 *= inv;
  unsigned short b0 = f2bf(c0), b1 = f2bf(c1);
  ctx0[b * ld0 + tid] = b0; ctx0[b * ld0 + tid + 128] = b1;
  ctx1[b * ld1 + tid] = b0; ctx1[b * ld1 + tid + 128] = b1;
}

// ---------------- host-side helpers ----------------
static inline void gemm(hipStream_t st, const unsigned short* A, int lda,
                        const unsigned short* W, const float* Cin, int ldcin,
                        float* Cout, int ldc, const float* bias, int permBias,
                        int M, int N, int K)
{
  int waves = (M >> 4) * (N >> 6);
  int grid = (waves * 32 + 255) / 256;
  k_gemm<<<grid, 256, 0, st>>>(A, lda, W, Cin, ldcin, Cout, ldc, bias, permBias, M, N, K);
}
static inline void gemm_lstm(hipStream_t st, const unsigned short* A, int lda,
                             const unsigned short* W, const float* Cin, int ldcin,
                             const float* bias, float* c,
                             unsigned short* h0t, int ld0,
                             unsigned short* h1t, int ld1,
                             float* hft, int ldf, int K)
{
  k_gemm_lstm<<<8, 256, 0, st>>>(A, lda, W, Cin, ldcin, bias, c,
                                 h0t, ld0, h1t, ld1, hft, ldf, K);
}
static inline void convT(hipStream_t st, const float* W, unsigned short* dst,
                         int K, int N, int ldk, int koff, int perm)
{
  int n = K * N;
  k_convT<<<(n + 255) / 256, 256, 0, st>>>(W, dst, K, N, ldk, koff, perm);
}
static inline void zero(hipStream_t st, void* p, size_t bytes)
{
  int n = (int)(bytes / 4);
  k_zero<<<(n + 255) / 256, 256, 0, st>>>((unsigned int*)p, n);
}

extern "C" void kernel_launch(void* const* d_in, const int* in_sizes, int n_in,
                              void* d_out, int out_size, void* d_ws, size_t ws_size,
                              hipStream_t stream)
{
  (void)in_sizes; (void)n_in; (void)out_size; (void)ws_size;
  const int*   src     = (const int*)d_in[0];
  const int*   tgt     = (const int*)d_in[1];
  const float* enc_emb = (const float*)d_in[2];
  const float* dec_emb = (const float*)d_in[3];
  const float* eWx0 = (const float*)d_in[4];
  const float* eWh0 = (const float*)d_in[5];
  const float* eb0  = (const float*)d_in[6];
  const float* eWx1 = (const float*)d_in[7];
  const float* eWh1 = (const float*)d_in[8];
  const float* eb1  = (const float*)d_in[9];
  const float* dWx0 = (const float*)d_in[10];
  const float* dWh0 = (const float*)d_in[11];
  const float* db0  = (const float*)d_in[12];
  const float* dWx1 = (const float*)d_in[13];
  const float* dWh1 = (const float*)d_in[14];
  const float* db1  = (const float*)d_in[15];
  const float* Wa   = (const float*)d_in[16];
  const float* Ua   = (const float*)d_in[17];
  const float* v_at = (const float*)d_in[18];
  const float* Wout = (const float*)d_in[19];
  const float* bout = (const float*)d_in[20];
  float* out = (float*)d_out;

  char* ws = (char*)d_ws; size_t off = 0;
  auto alloc = [&](size_t bytes) -> char* {
    char* p = ws + off; off = (off + bytes + 255) & ~(size_t)255; return p;
  };
  // bf16 weights (transposed [N][K]; recurrent ones gate-interleaved; K-concat fused)
  unsigned short* eWh0T = (unsigned short*)alloc(1024 * 256 * 2);
  unsigned short* eWx0T = (unsigned short*)alloc(1024 * 128 * 2);
  unsigned short* eW1T  = (unsigned short*)alloc(1024 * 512 * 2); // [Wx1 | Wh1]
  unsigned short* dW0T  = (unsigned short*)alloc(1024 * 640 * 2); // [Wx0(384) | Wh0(256)]
  unsigned short* dW1T  = (unsigned short*)alloc(1024 * 512 * 2); // [Wx1 | Wh1]
  unsigned short* WaT   = (unsigned short*)alloc(256 * 256 * 2);
  unsigned short* UaT   = (unsigned short*)alloc(256 * 256 * 2);
  unsigned short* WoT   = (unsigned short*)alloc(64 * 512 * 2);
  // bf16 activations
  unsigned short* embA  = (unsigned short*)alloc((size_t)SSRC * B_ * EMB_ * 2);
  unsigned short* eoBF  = (unsigned short*)alloc((size_t)B_ * SSRC * HID_ * 2);
  unsigned short* hcatE = (unsigned short*)alloc(B_ * 512 * 2);
  unsigned short* hcatD = (unsigned short*)alloc(B_ * 512 * 2);
  unsigned short* inp0  = (unsigned short*)alloc(B_ * 640 * 2);
  unsigned short* logA  = (unsigned short*)alloc(B_ * 512 * 2);
  // f32 buffers
  float* Xin0 = (float*)alloc((size_t)SSRC * B_ * G4 * 4);  // gate-permuted columns
  float* c0   = (float*)alloc(B_ * HID_ * 4);
  float* c1   = (float*)alloc(B_ * HID_ * 4);
  float* encO = (float*)alloc((size_t)B_ * SSRC * HID_ * 4);
  float* Uh   = (float*)alloc((size_t)B_ * SSRC * HID_ * 4);

  // ---- weight conversion (recurrent weights gate-interleaved) ----
  convT(stream, eWh0, eWh0T, 256, 1024, 256, 0, 1);
  convT(stream, eWx0, eWx0T, 128, 1024, 128, 0, 1);
  convT(stream, eWx1, eW1T,  256, 1024, 512, 0, 1);
  convT(stream, eWh1, eW1T,  256, 1024, 512, 256, 1);
  convT(stream, dWx0, dW0T,  384, 1024, 640, 0, 1);
  convT(stream, dWh0, dW0T,  256, 1024, 640, 384, 1);
  convT(stream, dWx1, dW1T,  256, 1024, 512, 0, 1);
  convT(stream, dWh1, dW1T,  256, 1024, 512, 256, 1);
  convT(stream, Wa,   WaT,   256, 256,  256, 0, 0);
  convT(stream, Ua,   UaT,   256, 256,  256, 0, 0);
  convT(stream, Wout, WoT,   512, 64,   512, 0, 0);

  // ---- zero initial state ----
  zero(stream, c0, B_ * HID_ * 4);
  zero(stream, c1, B_ * HID_ * 4);
  zero(stream, hcatE, B_ * 512 * 2);

  // ---- hoisted encoder input projection: Xin0 = emb@Wx0 + b0 (permuted cols) ----
  {
    int n = SSRC * B_ * EMB_;
    k_gather_src<<<(n + 255) / 256, 256, 0, stream>>>(src, enc_emb, embA);
  }
  gemm(stream, embA, EMB_, eWx0T, nullptr, 0, Xin0, G4, eb0, 1, SSRC * B_, G4, EMB_);

  // ---- encoder recurrence: one fused GEMM+cell per layer per step ----
  for (int t = 0; t < SSRC; ++t) {
    gemm_lstm(stream, hcatE, 512, eWh0T, Xin0 + (size_t)t * B_ * G4, G4,
              nullptr, c0, hcatE, 512, nullptr, 0, nullptr, 0, HID_);
    gemm_lstm(stream, hcatE, 512, eW1T, nullptr, 0,
              eb1, c1, hcatE + 256, 512, eoBF + (size_t)t * HID_, SSRC * HID_,
              encO + (size_t)t * HID_, SSRC * HID_, 512);
  }

  // ---- Uh = enc_out @ Ua (parallel) ----
  gemm(stream, eoBF, HID_, UaT, nullptr, 0, Uh, HID_, nullptr, 0, B_ * SSRC, HID_, HID_);

  // ---- decoder initial state = encoder final state ----
  {
    int n = 4 * B_ * HID_;
    k_copy_dec_init<<<(n + 255) / 256, 256, 0, stream>>>(hcatE, hcatD, inp0, logA);
  }

  // ---- decoder recurrence (teacher forcing, 63 steps, 4 launches/step) ----
  for (int t = 0; t < STGT - 1; ++t) {
    k_attn_fused<<<B_, 128, 0, stream>>>(hcatD, WaT, Uh, v_at, encO,
                                         tgt, dec_emb, t,
                                         inp0 + 128, 640, logA + 256, 512, inp0);
    gemm_lstm(stream, inp0, 640, dW0T, nullptr, 0,
              db0, c0, inp0 + 384, 640, hcatD, 512, nullptr, 0, 640);
    gemm_lstm(stream, hcatD, 512, dW1T, nullptr, 0,
              db1, c1, hcatD + 256, 512, logA, 512, nullptr, 0, 512);
    gemm(stream, logA, 512, WoT, nullptr, 0, out + (size_t)t * VTGT,
         (STGT - 1) * VTGT, bout, 0, B_, VTGT, 512);
  }
}
